// BasicBlock1_1w4a_A_13804024889723
// MI455X (gfx1250) — 8-bit matrix operand order repaired and checked by reading, compiled, not run on hardware
//
#include <hip/hip_runtime.h>
#include <hip/hip_bf16.h>
#include <stdint.h>

#define NB   32
#define CIN  64
#define COUT 64
#define H    112
#define W    112
#define HW   (H*W)
#define NPIX (NB*HW)
#define WELEMS (COUT*CIN*9)
#define A_LEVELS 15.0f
#define BN_EPS   1e-5f

#define PXSTRIDE  80
#define ROWPX     114
#define ROWSTRIDE (ROWPX*PXSTRIDE)

typedef int v8i __attribute__((ext_vector_type(8)));

__global__ __launch_bounds__(256) void k_alpha(const float* __restrict__ w,
                                               float* __restrict__ alpha) {
  __shared__ float red[256];
  float s = 0.f;
  for (int i = threadIdx.x; i < WELEMS; i += 256) s += fabsf(w[i]);
  red[threadIdx.x] = s;
  __syncthreads();
  for (int o = 128; o > 0; o >>= 1) {
    if ((int)threadIdx.x < o) red[threadIdx.x] += red[threadIdx.x + o];
    __syncthreads();
  }
  if (threadIdx.x == 0) *alpha = red[0] / (float)WELEMS;
}

__global__ void k_scale_bias(const float* __restrict__ gamma,
                             const float* __restrict__ beta,
                             const float* __restrict__ bn_mean,
                             const float* __restrict__ bn_var,
                             const float* __restrict__ alpha,
                             float* __restrict__ scale,
                             float* __restrict__ bias) {
  int c = threadIdx.x;
  if (c >= COUT) return;
  float inv = gamma[c] / sqrtf(bn_var[c] + BN_EPS);
  scale[c] = inv * (*alpha) / A_LEVELS;
  bias[c]  = beta[c] - bn_mean[c] * inv;
}

__global__ __launch_bounds__(256) void k_pack_w(const float* __restrict__ w,
                                                int* __restrict__ pw) {
  int idx = blockIdx.x * 256 + threadIdx.x;
  if (idx >= 4 * 9 * 32 * 8) return;
  int d    = idx & 7;
  int lane = (idx >> 3) & 31;
  int tm   = idx >> 8;
  int tap  = tm % 9;
  int mt   = tm / 9;
  int m    = lane & 15;
  int half = lane >> 4;
  int kh   = tap / 3, kw = tap % 3;
  unsigned int dword = 0;
  for (int b = 0; b < 4; ++b) {
    int cin  = ((d >> 2) << 5) + ((d & 3) << 2) + (half << 4) + b;
    int cout = mt * 16 + m;
    float wv = w[((cout * CIN + cin) * 3 + kh) * 3 + kw];
    int s = (wv > 0.f) ? 1 : ((wv < 0.f) ? -1 : 0);
    dword |= ((unsigned int)(s & 0xFF)) << (8 * b);
  }
  pw[idx] = (int)dword;
}

__global__ __launch_bounds__(256) void k_quant(const float* __restrict__ x,
                                               unsigned char* __restrict__ q) {
  __shared__ __align__(16) unsigned char tile[64 * 80];
  int blk = blockIdx.x;
  int n = blk / 196, pt = blk % 196;
  int pbase = pt * 64;
  const float* xin = x + (size_t)n * CIN * HW + pbase;
  for (int e = threadIdx.x; e < 4096; e += 256) {
    int c = e >> 6, p = e & 63;
    float v = xin[(size_t)c * HW + p];
    v = fminf(fmaxf(v, 0.f), 1.f);
    unsigned int qi = (unsigned int)(v * A_LEVELS + 0.5f);
    tile[p * 80 + c] = (unsigned char)qi;
  }
  __syncthreads();
  int p = threadIdx.x >> 2, ch = threadIdx.x & 3;
  uint4 vv = *(const uint4*)&tile[p * 80 + ch * 16];
  *(uint4*)&q[(size_t)(n * HW + pbase + p) * 64 + ch * 16] = vv;
}

__global__ __launch_bounds__(224) void k_conv_wmma(const unsigned char* __restrict__ qact,
                                                   const int* __restrict__ pw,
                                                   const float* __restrict__ scale,
                                                   const float* __restrict__ bias,
                                                   float* __restrict__ out) {
  __shared__ __align__(32) int wlds[4 * 9 * 32 * 8];
  __shared__ __align__(16) unsigned char acts[3 * ROWSTRIDE];

  const int tid = threadIdx.x;

  for (int i = tid; i < 4 * 9 * 32 * 8; i += 224) wlds[i] = pw[i];

  const int rowId = blockIdx.x;
  const int n  = rowId / H;
  const int oh = rowId - n * H;

  const uint4 z = {0u, 0u, 0u, 0u};
  if (tid < 24) {
    int r  = tid >> 3;
    int h8 = tid & 7;
    int px = (h8 >> 2) ? (ROWPX - 1) : 0;
    int ch = h8 & 3;
    *(uint4*)&acts[r * ROWSTRIDE + px * PXSTRIDE + ch * 16] = z;
  }

  for (int r = 0; r < 3; ++r) {
    const int ih = oh + r - 1;
    if ((unsigned)ih >= (unsigned)H) {
      for (int c = tid; c < ROWSTRIDE / 16; c += 224)
        *(uint4*)&acts[r * ROWSTRIDE + c * 16] = z;
    } else {
      const unsigned char* rowp = qact + (size_t)((n * H + ih) * W) * 64;
      for (int c = tid; c < (W * 64) / 16; c += 224) {
        int px = c >> 2, ch = c & 3;
        unsigned int lo = (unsigned int)(uintptr_t)
            &acts[r * ROWSTRIDE + (px + 1) * PXSTRIDE + ch * 16];
        unsigned long long ga = (unsigned long long)(uintptr_t)(rowp + c * 16);
        asm volatile("global_load_async_to_lds_b128 %0, %1, off"
                     :: "v"(lo), "v"(ga) : "memory");
      }
    }
  }

  asm volatile("s_wait_asynccnt 0x0" ::: "memory");
  __syncthreads();

  const int wave = tid >> 5;
  const int lane = tid & 31;
  const int col  = lane & 15;
  const int half = lane >> 4;
  const int owBase = wave * 16;

  v8i acc0 = {}, acc1 = {}, acc2 = {}, acc3 = {};
#pragma unroll
  for (int kh = 0; kh < 3; ++kh) {
#pragma unroll
    for (int kw = 0; kw < 3; ++kw) {
      const int tap = kh * 3 + kw;
      const unsigned char* bp =
          &acts[kh * ROWSTRIDE + (owBase + col + kw) * PXSTRIDE + half * 16];
      int4 b0 = *(const int4*)(bp);
      int4 b1 = *(const int4*)(bp + 32);
      v8i b;
      b[0] = b0.x; b[1] = b0.y; b[2] = b0.z; b[3] = b0.w;
      b[4] = b1.x; b[5] = b1.y; b[6] = b1.z; b[7] = b1.w;

      const v8i a0 = *(const v8i*)&wlds[((0 * 9 + tap) * 32 + lane) * 8];
      acc0 = __builtin_amdgcn_wmma_i32_16x16x64_iu8(true, a0, false, b, acc0, false, false);
      const v8i a1 = *(const v8i*)&wlds[((1 * 9 + tap) * 32 + lane) * 8];
      acc1 = __builtin_amdgcn_wmma_i32_16x16x64_iu8(true, a1, false, b, acc1, false, false);
      const v8i a2 = *(const v8i*)&wlds[((2 * 9 + tap) * 32 + lane) * 8];
      acc2 = __builtin_amdgcn_wmma_i32_16x16x64_iu8(true, a2, false, b, acc2, false, false);
      const v8i a3 = *(const v8i*)&wlds[((3 * 9 + tap) * 32 + lane) * 8];
      acc3 = __builtin_amdgcn_wmma_i32_16x16x64_iu8(true, a3, false, b, acc3, false, false);
    }
  }

  const int ow = owBase + col;
#pragma unroll
  for (int r = 0; r < 8; ++r) {
    const int coL = r + (half << 3);
    {
      const int co = 0 * 16 + coL;
      float v = fmaxf((float)acc0[r] * scale[co] + bias[co], 0.f);
      out[((size_t)n * COUT + co) * HW + oh * W + ow] = v;
    }
    {
      const int co = 1 * 16 + coL;
      float v = fmaxf((float)acc1[r] * scale[co] + bias[co], 0.f);
      out[((size_t)n * COUT + co) * HW + oh * W + ow] = v;
    }
    {
      const int co = 2 * 16 + coL;
      float v = fmaxf((float)acc2[r] * scale[co] + bias[co], 0.f);
      out[((size_t)n * COUT + co) * HW + oh * W + ow] = v;
    }
    {
      const int co = 3 * 16 + coL;
      float v = fmaxf((float)acc3[r] * scale[co] + bias[co], 0.f);
      out[((size_t)n * COUT + co) * HW + oh * W + ow] = v;
    }
  }
}

extern "C" void kernel_launch(void* const* d_in, const int* in_sizes, int n_in,
                              void* d_out, int out_size, void* d_ws, size_t ws_size,
                              hipStream_t stream) {
  const float* x       = (const float*)d_in[0];
  const float* w       = (const float*)d_in[1];
  const float* gamma   = (const float*)d_in[2];
  const float* beta    = (const float*)d_in[3];
  const float* bn_mean = (const float*)d_in[4];
  const float* bn_var  = (const float*)d_in[5];
  float* out = (float*)d_out;

  const size_t QACT_BYTES = (size_t)NPIX * CIN;
  unsigned char* qact = (unsigned char*)d_ws;
  int*   pw    = (int*)  ((char*)d_ws + QACT_BYTES);
  float* alpha = (float*)((char*)d_ws + QACT_BYTES + 36864);
  float* scale = (float*)((char*)d_ws + QACT_BYTES + 36864 + 256);
  float* bias  = scale + 64;

  k_alpha<<<1, 256, 0, stream>>>(w, alpha);
  k_scale_bias<<<1, 64, 0, stream>>>(gamma, beta, bn_mean, bn_var, alpha, scale, bias);
  k_pack_w<<<36, 256, 0, stream>>>(w, pw);
  k_quant<<<NB * (HW / 64), 256, 0, stream>>>(x, qact);

  k_conv_wmma<<<NB * H, 224, 0, stream>>>(qact, pw, scale, bias, out);
}
